// Reservoir_75024488727228
// MI455X (gfx1250) — matrix operand order repaired and checked by reading, not compiled, not run on hardware
//
#include <hip/hip_runtime.h>

// ---------------- problem constants ----------------
#define B_    32
#define T_    2000
#define FIN_  256
#define H_    1024
#define OUT_  256
#define GG    1.2f
#define DTAU  0.1f     // DT / TAU

// ---------------- launch geometry ------------------
#define NWG      32     // persistent workgroups
#define HS       32     // hidden columns owned per WG
#define NTHREADS 128    // 4 waves of 32 (wave32)
#define NWAVES   4

typedef __attribute__((ext_vector_type(16))) __bf16 v16bf;
typedef __attribute__((ext_vector_type(8)))  float  v8f;

union ABfrag { v16bf v; uint4 q[2]; unsigned u[8]; };

__device__ __forceinline__ unsigned bf16r(float f) {
  unsigned u = __float_as_uint(f);
  return (u + 0x7FFFu + ((u >> 16) & 1u)) >> 16;   // round-to-nearest-even
}
__device__ __forceinline__ unsigned short bf16h(float f) { return (unsigned short)bf16r(f); }
__device__ __forceinline__ unsigned packbf(float lo, float hi) { return bf16r(lo) | (bf16r(hi) << 16); }

// A-tile (16x32 bf16, MxK) from row-major bf16 [rows][ld]
__device__ __forceinline__ v16bf load_a16(const unsigned short* __restrict__ src,
                                          int ld, int m0, int k0, int lane) {
  int half = lane >> 4;
  const uint4* p = (const uint4*)(src + (size_t)(m0 + (lane & 15)) * ld + k0 + half * 8);
  ABfrag a; a.q[0] = p[0]; a.q[1] = p[2];   // K {0..7} and K {16..23} of this half
  return a.v;
}

// B-tile (32x16 bf16, KxN) from K-contiguous transposed weights WT[n][k], row stride ldk
__device__ __forceinline__ v16bf load_bt16(const unsigned short* __restrict__ wt,
                                           int ldk, int k0, int n0, int lane) {
  int koff = (lane >> 4) << 3;              // lanes 0-15: K 0..7,16..23; lanes 16-31: K 8..15,24..31
  const uint4* p = (const uint4*)(wt + (size_t)(n0 + (lane & 15)) * ldk + k0 + koff);
  ABfrag b; b.q[0] = p[0]; b.q[1] = p[2];
  return b.v;
}

// A-tile from fp32 source, converting to bf16 on the fly
__device__ __forceinline__ v16bf load_a32(const float* __restrict__ src,
                                          long ld, int m0, int k0, int lane) {
  int half = lane >> 4;
  const float4* p = (const float4*)(src + (long)(m0 + (lane & 15)) * ld + k0 + half * 8);
  float4 f0 = p[0], f1 = p[1], f2 = p[4], f3 = p[5];
  ABfrag a;
  a.u[0] = packbf(f0.x, f0.y); a.u[1] = packbf(f0.z, f0.w);
  a.u[2] = packbf(f1.x, f1.y); a.u[3] = packbf(f1.z, f1.w);
  a.u[4] = packbf(f2.x, f2.y); a.u[5] = packbf(f2.z, f2.w);
  a.u[6] = packbf(f3.x, f3.y); a.u[7] = packbf(f3.z, f3.w);
  return a.v;
}

__device__ __forceinline__ v8f wmma_bf16(v16bf a, v16bf b, v8f c) {
  return __builtin_amdgcn_wmma_f32_16x16x32_bf16(false, a, false, b, (short)0, c, false, false);
}

// Monotonic grid-wide barrier (counter zeroed per launch by prep kernel)
__device__ __forceinline__ void grid_barrier(unsigned* bar, unsigned gen) {
  __threadfence();
  __syncthreads();
  if (threadIdx.x == 0) {
    atomicAdd(bar, 1u);
    unsigned target = gen * NWG;
    while (__atomic_load_n(bar, __ATOMIC_RELAXED) < target)
      __builtin_amdgcn_s_sleep(2);
  }
  __syncthreads();
  __threadfence();
}

__global__ void __launch_bounds__(NTHREADS, 1)
reservoir_main(const float* __restrict__ inputs,          // [B][T][FIN]
               const float* __restrict__ xih,             // [B][H]
               const float* __restrict__ xio,             // [B][OUT]
               const unsigned short* __restrict__ WeffT,  // bf16 [H(out)][H(k)] = (G*Wr + Wo@Wf)^T
               const unsigned short* __restrict__ WiT,    // bf16 [H][FIN]
               const unsigned short* __restrict__ WoT,    // bf16 [OUT][H]
               const float* __restrict__ biasF,           // fp32 [B][H] = xi_out @ Wf
               unsigned short* __restrict__ rbuf,         // bf16 [2][B][H]
               float* __restrict__ out,                   // [B][T][OUT]
               unsigned* __restrict__ bar) {
  __shared__ unsigned short sWe[HS * H_];     // 64 KB: Weff cols [hbase,hbase+32)
  __shared__ unsigned short sWi[HS * FIN_];   // 16 KB
  __shared__ unsigned short sWo[16 * H_];     // 32 KB: Wo cols [zbase,zbase+16)
  __shared__ float sZ[NWAVES * 256];          //  4 KB: z K-split reduction

  const int g     = blockIdx.x;
  const int tid   = threadIdx.x;
  const int wave  = tid >> 5;
  const int lane  = tid & 31;
  const int hbase = g * HS;                   // hidden column slice
  const int zr    = g & 1;                    // z row-tile (batch 0-15 / 16-31)
  const int zbase = (g >> 1) * 16;            // z column slice

  // ---- park weight slices in LDS (b128 copies) ----
  {
    const uint4* s; uint4* d; int n;
    s = (const uint4*)(WeffT + (size_t)hbase * H_); d = (uint4*)sWe; n = HS * H_ / 8;
    for (int i = tid; i < n; i += NTHREADS) d[i] = s[i];
    s = (const uint4*)(WiT + (size_t)hbase * FIN_); d = (uint4*)sWi; n = HS * FIN_ / 8;
    for (int i = tid; i < n; i += NTHREADS) d[i] = s[i];
    s = (const uint4*)(WoT + (size_t)zbase * H_);   d = (uint4*)sWo; n = 16 * H_ / 8;
    for (int i = tid; i < n; i += NTHREADS) d[i] = s[i];
  }

  // per-wave du tile: rows m0..m0+15 (batch), cols hbase+ncl..+15 (hidden)
  const int m0   = (wave & 1) * 16;
  const int ncl  = (wave >> 1) * 16;
  const int half = lane >> 4;
  const int nlan = lane & 15;

  v8f u = {};                     // hidden state tile, resident in accumulator VGPRs
  float xh[8], xo[8], bs[8];
  #pragma unroll
  for (int v = 0; v < 8; ++v) {
    size_t idx = (size_t)(m0 + v + 8 * half) * H_ + hbase + ncl + nlan;
    xh[v] = xih[idx];
    bs[v] = biasF[idx];           // time-invariant xi_out@Wf term, kept in VGPRs
  }
  #pragma unroll
  for (int v = 0; v < 8; ++v)     // only wave 0 consumes xo
    xo[v] = xio[(size_t)(16 * zr + v + 8 * half) * OUT_ + zbase + nlan];

  __syncthreads();

  // r_0 = tanh(0) + xi_hidden = xi_hidden  -> parity 0
  #pragma unroll
  for (int v = 0; v < 8; ++v)
    rbuf[(size_t)(m0 + v + 8 * half) * H_ + hbase + ncl + nlan] = bf16h(xh[v]);

  unsigned gen = 0;
  grid_barrier(bar, ++gen);

  for (int t = 0; t < T_; ++t) {
    const unsigned short* rp = rbuf + (size_t)(t & 1) * (B_ * H_);
    unsigned short*       rn = rbuf + (size_t)((t + 1) & 1) * (B_ * H_);

    // prefetch next timestep's input rows (global_prefetch_b8)
    if (t + 1 < T_) {
      const float* nx = inputs + (size_t)(m0 + nlan) * ((size_t)T_ * FIN_)
                               + (size_t)(t + 1) * FIN_ + half * 128;
      __builtin_prefetch(nx, 0, 1);
    }

    // ---- Z phase (output only): z(:, zbase:+16) = r @ Wo + xi_out; K split over 4 waves
    v8f zacc = {};
    #pragma unroll 4
    for (int kk = 0; kk < 8; ++kk) {
      int k = wave * 256 + kk * 32;
      zacc = wmma_bf16(load_a16(rp, H_, zr * 16, k, lane),
                       load_bt16(sWo, H_, k, 0, lane), zacc);
    }
    #pragma unroll
    for (int v = 0; v < 8; ++v)
      sZ[wave * 256 + (v + 8 * half) * 16 + nlan] = zacc[v];
    __syncthreads();
    if (wave == 0) {
      #pragma unroll
      for (int v = 0; v < 8; ++v) {
        int off = (v + 8 * half) * 16 + nlan;
        float s = sZ[off] + sZ[256 + off] + sZ[512 + off] + sZ[768 + off] + xo[v];
        int brow = 16 * zr + v + 8 * half;
        out[(size_t)brow * ((size_t)T_ * OUT_) + (size_t)t * OUT_ + zbase + nlan] = s;
      }
    }

    // ---- DU phase: du = r@Weff + x_t@Wi + bias   (z feedback folded into Weff/bias)
    v8f acc = {};
    #pragma unroll 4
    for (int k = 0; k < H_; k += 32)
      acc = wmma_bf16(load_a16(rp, H_, m0, k, lane),
                      load_bt16(sWe, H_, k, ncl, lane), acc);
    #pragma unroll 4
    for (int k = 0; k < FIN_; k += 32)
      acc = wmma_bf16(load_a32(inputs + (size_t)t * FIN_, (long)T_ * FIN_, m0, k, lane),
                      load_bt16(sWi, FIN_, k, ncl, lane), acc);

    #pragma unroll
    for (int v = 0; v < 8; ++v) {
      float du = acc[v] + bs[v];
      float un = u[v] + DTAU * (du - u[v]);
      u[v] = un;
      float rv = tanhf(un) + xh[v];
      rn[(size_t)(m0 + v + 8 * half) * H_ + hbase + ncl + nlan] = bf16h(rv);
    }
    grid_barrier(bar, ++gen);     // r_{t+1} visible device-wide (only barrier per step)
  }
}

// ---- prep: W[K][N] fp32 -> WT[N][K] bf16 (K-contiguous for B-fragments) ----
__global__ void k_transpose_bf16(const float* __restrict__ W,
                                 unsigned short* __restrict__ WT, int K, int N) {
  long idx = (long)blockIdx.x * 256 + threadIdx.x;
  if (idx >= (long)K * N) return;
  int n = (int)(idx / K);
  int k = (int)(idx % K);
  WT[idx] = bf16h(W[(long)k * N + n]);
}

// ---- prep: WeffT[n][k] = bf16( G*Wr[k][n] + sum_j Wo[k][j]*Wf[j][n] ); one block per k
__global__ void __launch_bounds__(256)
k_weff(const float* __restrict__ Wr, const float* __restrict__ Wo,
       const float* __restrict__ Wf, unsigned short* __restrict__ WeffT) {
  __shared__ float srow[OUT_];
  const int k = blockIdx.x;
  for (int j = threadIdx.x; j < OUT_; j += 256) srow[j] = Wo[(size_t)k * OUT_ + j];
  __syncthreads();
  for (int n = threadIdx.x; n < H_; n += 256) {
    float acc = GG * Wr[(size_t)k * H_ + n];
    #pragma unroll 4
    for (int j = 0; j < OUT_; ++j) acc += srow[j] * Wf[(size_t)j * H_ + n];
    WeffT[(size_t)n * H_ + k] = bf16h(acc);
  }
}

// ---- prep: bias[b][n] = sum_j xi_out[b][j]*Wf[j][n]; one block per b
__global__ void __launch_bounds__(256)
k_bias(const float* __restrict__ xio, const float* __restrict__ Wf,
       float* __restrict__ bias) {
  __shared__ float srow[OUT_];
  const int b = blockIdx.x;
  for (int j = threadIdx.x; j < OUT_; j += 256) srow[j] = xio[(size_t)b * OUT_ + j];
  __syncthreads();
  for (int n = threadIdx.x; n < H_; n += 256) {
    float acc = 0.f;
    #pragma unroll 4
    for (int j = 0; j < OUT_; ++j) acc += srow[j] * Wf[(size_t)j * H_ + n];
    bias[(size_t)b * H_ + n] = acc;
  }
}

__global__ void k_zero_u32(unsigned* p, int n) {
  int i = blockIdx.x * 256 + threadIdx.x;
  if (i < n) p[i] = 0u;
}

extern "C" void kernel_launch(void* const* d_in, const int* in_sizes, int n_in,
                              void* d_out, int out_size, void* d_ws, size_t ws_size,
                              hipStream_t stream) {
  const float* inputs = (const float*)d_in[0];
  const float* Wr     = (const float*)d_in[1];
  const float* Wi     = (const float*)d_in[2];
  const float* Wf     = (const float*)d_in[3];
  const float* Wo     = (const float*)d_in[4];
  const float* xih    = (const float*)d_in[5];
  const float* xio    = (const float*)d_in[6];
  float* out = (float*)d_out;

  char* ws = (char*)d_ws;
  size_t off = 0;
  unsigned short* WeffT = (unsigned short*)(ws + off); off += (size_t)H_   * H_   * 2;
  unsigned short* WiT   = (unsigned short*)(ws + off); off += (size_t)H_   * FIN_ * 2;
  unsigned short* WoT   = (unsigned short*)(ws + off); off += (size_t)OUT_ * H_   * 2;
  float*          biasF = (float*)(ws + off);          off += (size_t)B_   * H_   * 4;
  unsigned short* rbuf  = (unsigned short*)(ws + off); off += (size_t)2 * B_ * H_ * 2;
  off = (off + 255) & ~(size_t)255;
  unsigned* bar = (unsigned*)(ws + off);

  k_weff<<<H_, 256, 0, stream>>>(Wr, Wo, Wf, WeffT);
  k_transpose_bf16<<<(FIN_ * H_ + 255) / 256, 256, 0, stream>>>(Wi, WiT, FIN_, H_);
  k_transpose_bf16<<<(H_ * OUT_ + 255) / 256, 256, 0, stream>>>(Wo, WoT, H_, OUT_);
  k_bias<<<B_, 256, 0, stream>>>(xio, Wf, biasF);
  k_zero_u32<<<1, 256, 0, stream>>>(bar, 1);

  reservoir_main<<<NWG, NTHREADS, 0, stream>>>(inputs, xih, xio, WeffT, WiT, WoT, biasF,
                                               rbuf, out, bar);
}
